// SparseEvoFormerBlock_75282186764800
// MI455X (gfx1250) — matrix operand order repaired and checked by reading, not compiled, not run on hardware
//
#include <hip/hip_runtime.h>
#include <hip/hip_bf16.h>

// ---------------------------------------------------------------------------
// Evoformer block for MI455X (gfx1250), wave32 + WMMA bf16 (fp32 accumulate).
// GEMMs: LDS-staged 64x64 block tiles, ds_load_b128 fragments, WMMA math.
// Attention: fully fused per (batch,head) using CDNA5's 320KB LDS
// (48KB Q/K/Vt + 256KB score matrix resident, zero logits HBM traffic).
// ---------------------------------------------------------------------------

typedef __attribute__((ext_vector_type(16))) __bf16          v16bf;
typedef __attribute__((ext_vector_type(16))) unsigned short  v16us;
typedef __attribute__((ext_vector_type(8)))  float           v8f;

union Frag { v16us v; uint4 q[2]; };

__device__ __forceinline__ unsigned short f2bfu(float f) {
  unsigned int u = __builtin_bit_cast(unsigned int, f);
  u += 0x7fffu + ((u >> 16) & 1u);               // round-to-nearest-even
  return (unsigned short)(u >> 16);
}

// pack 8 contiguous f32 (32B-aligned) -> 8 bf16 in a uint4
__device__ __forceinline__ uint4 pack8(const float* p) {
  const float4 f0 = *(const float4*)p;
  const float4 f1 = *(const float4*)(p + 4);
  uint4 r;
  r.x = (unsigned)f2bfu(f0.x) | ((unsigned)f2bfu(f0.y) << 16);
  r.y = (unsigned)f2bfu(f0.z) | ((unsigned)f2bfu(f0.w) << 16);
  r.z = (unsigned)f2bfu(f1.x) | ((unsigned)f2bfu(f1.y) << 16);
  r.w = (unsigned)f2bfu(f1.z) | ((unsigned)f2bfu(f1.w) << 16);
  return r;
}

__device__ __forceinline__ v8f wmma_bf16(v16us a, v16us b, v8f c) {
  return __builtin_amdgcn_wmma_f32_16x16x32_bf16(
      false, __builtin_bit_cast(v16bf, a),
      false, __builtin_bit_cast(v16bf, b),
      (short)0, c, false, false);
}

// ---------------------------------------------------------------------------
// GEMM: Out[M,N] = act(A[M,K] @ W[K,N] + bias[N]) (+ Res[M,N])
// Block tile 64x64; 8 waves as 4(m) x 2(n); each wave: 16x32 via 2 WMMA accs
// sharing one A fragment. A staged [m][k], W staged transposed [n][k] so all
// fragments are aligned b128 LDS loads. M,K mult of 64/32; N mult of 64.
// ---------------------------------------------------------------------------
template<bool ABF16, bool OBF16, bool RELU, bool RES>
__global__ __launch_bounds__(256)
void gemm_kernel(const void* __restrict__ Ap, const float* __restrict__ W,
                 const float* __restrict__ bias, const float* __restrict__ Resid,
                 void* __restrict__ Outp, int M, int N, int K) {
  __shared__ __align__(16) unsigned short As[64 * 32];  // [mrow][k]
  __shared__ __align__(16) unsigned short Bs[64 * 32];  // [ncol][k] (transposed)
  const int nb = N >> 6;
  const int bm = blockIdx.x / nb, bn = blockIdx.x % nb;
  const int m0 = bm << 6, n0 = bn << 6;
  const int tid = threadIdx.x;
  const int wv = tid >> 5, lane = tid & 31;
  const int wm = wv >> 1, wn = wv & 1;                  // wave coords: 4 x 2
  const int lo = lane & 15, hi = lane >> 4;
  // staging coords
  const int ar = tid >> 2, ac = (tid & 3) << 3;         // A: 64 rows x 4 chunks
  const int bk = tid >> 3, bc = (tid & 7) << 3;         // W: 32 k-rows x 8 chunks
  const float*          Af = (const float*)Ap;
  const unsigned short* Ab = (const unsigned short*)Ap;
  v8f acc0 = {}, acc1 = {};

  for (int kb = 0; kb < K; kb += 32) {
    // ---- stage A tile (64x32) ----
    const long gA = (long)(m0 + ar) * K + kb + ac;
    if (ABF16) {
      *(uint4*)&As[ar * 32 + ac] = *(const uint4*)(Ab + gA);
    } else {
      *(uint4*)&As[ar * 32 + ac] = pack8(Af + gA);
    }
    // ---- stage W tile (32x64), transposed into Bs[n][k] ----
    const long gB = (long)(kb + bk) * N + n0 + bc;
    {
      const float4 f0 = *(const float4*)(W + gB);
      const float4 f1 = *(const float4*)(W + gB + 4);
      const float fv[8] = {f0.x, f0.y, f0.z, f0.w, f1.x, f1.y, f1.z, f1.w};
      #pragma unroll
      for (int t = 0; t < 8; ++t) Bs[(bc + t) * 32 + bk] = f2bfu(fv[t]);
    }
    if (kb + 32 < K) {                                  // prefetch next k-step
      __builtin_prefetch(ABF16 ? (const void*)(Ab + gA + 32)
                               : (const void*)(Af + gA + 32), 0, 1);
      __builtin_prefetch((const void*)(W + gB + 32L * N), 0, 1);
    }
    __syncthreads();
    // ---- fragments: aligned b128 LDS loads ----
    const int arow = wm * 16 + lo;
    Frag a, b0, b1;
    a.q[0] = *(const uint4*)&As[arow * 32 + hi * 8];
    a.q[1] = *(const uint4*)&As[arow * 32 + 16 + hi * 8];
    const int c0 = wn * 32 + lo, c1 = c0 + 16;
    b0.q[0] = *(const uint4*)&Bs[c0 * 32 + hi * 8];
    b0.q[1] = *(const uint4*)&Bs[c0 * 32 + 16 + hi * 8];
    b1.q[0] = *(const uint4*)&Bs[c1 * 32 + hi * 8];
    b1.q[1] = *(const uint4*)&Bs[c1 * 32 + 16 + hi * 8];
    acc0 = wmma_bf16(a.v, b0.v, acc0);
    acc1 = wmma_bf16(a.v, b1.v, acc1);
    __syncthreads();
  }
  #pragma unroll
  for (int j = 0; j < 8; ++j) {
    const int r = m0 + wm * 16 + hi * 8 + j;
    const int c = n0 + wn * 32 + lo;
    float v0 = acc0[j] + bias[c];
    float v1 = acc1[j] + bias[c + 16];
    if (RELU) { v0 = fmaxf(v0, 0.0f); v1 = fmaxf(v1, 0.0f); }
    const long o0 = (long)r * N + c, o1 = o0 + 16;
    if (RES) { v0 += Resid[o0]; v1 += Resid[o1]; }
    if (OBF16) {
      ((unsigned short*)Outp)[o0] = f2bfu(v0);
      ((unsigned short*)Outp)[o1] = f2bfu(v1);
    } else {
      ((float*)Outp)[o0] = v0;
      ((float*)Outp)[o1] = v1;
    }
  }
}

// ---------------------------------------------------------------------------
// Fused attention for one (batch, head). Q/K staged [row][d] (bf16), V staged
// transposed [d][row], 256x256 logits resident in LDS. Element (row,d) of
// Q/K/V/O lives at base + row*RS + d, base = batch*BS + head*32.
// ---------------------------------------------------------------------------
template<bool HAS_BIAS>
__global__ __launch_bounds__(256)
void attn_kernel(const unsigned short* __restrict__ Q,
                 const unsigned short* __restrict__ Km,
                 const unsigned short* __restrict__ V,
                 unsigned short* __restrict__ O,
                 const float* __restrict__ bias, int nh, int BS, int RS) {
  extern __shared__ __align__(16) char smem[];
  unsigned short* Qs = (unsigned short*)smem;           // [256][32] 16KB
  unsigned short* Ks = Qs + 256 * 32;                   // [256][32] 16KB
  unsigned short* Vt = Ks + 256 * 32;                   // [32][256] 16KB
  float*          Ls = (float*)(Vt + 32 * 256);         // [256][256] 256KB

  const int tid   = threadIdx.x;
  const int batch = blockIdx.x / nh;
  const int h     = blockIdx.x % nh;
  const int base  = batch * BS + h * 32;

  for (int idx = tid; idx < 1024; idx += 256) {         // 256 rows x 4 chunks
    const int r = idx >> 2, db = (idx & 3) << 3;
    const int gi = base + r * RS + db;
    *(uint4*)&Qs[r * 32 + db] = *(const uint4*)(Q + gi);
    *(uint4*)&Ks[r * 32 + db] = *(const uint4*)(Km + gi);
    const uint4 vq = *(const uint4*)(V + gi);
    const unsigned short* vs = (const unsigned short*)&vq;
    #pragma unroll
    for (int t = 0; t < 8; ++t) Vt[(db + t) * 256 + r] = vs[t];
  }
  __syncthreads();

  const int wv = tid >> 5, lane = tid & 31;
  const int lo = lane & 15, hi = lane >> 4;
  const float scale = 0.17677669529663687f;             // 1/sqrt(32)

  // ---- logits: A fragment hoisted per q-tile, 16 k-tiles each ----
  for (int tq = wv; tq < 16; tq += 8) {
    const int qrow = tq * 16 + lo;
    Frag a;
    a.q[0] = *(const uint4*)&Qs[qrow * 32 + hi * 8];
    a.q[1] = *(const uint4*)&Qs[qrow * 32 + 16 + hi * 8];
    for (int tk = 0; tk < 16; ++tk) {
      const int kcol = tk * 16 + lo;
      Frag b;
      b.q[0] = *(const uint4*)&Ks[kcol * 32 + hi * 8];
      b.q[1] = *(const uint4*)&Ks[kcol * 32 + 16 + hi * 8];
      v8f acc = {};
      acc = wmma_bf16(a.v, b.v, acc);
      #pragma unroll
      for (int j = 0; j < 8; ++j) {
        const int r = tq * 16 + hi * 8 + j;
        float f = acc[j] * scale;
        if (HAS_BIAS) f += bias[(h * 256 + r) * 256 + kcol];
        Ls[r * 256 + kcol] = f;
      }
    }
  }
  __syncthreads();

  // ---- row softmax: each lane owns a contiguous 8-float chunk ----
  for (int r = wv; r < 256; r += 8) {
    float* row = Ls + r * 256 + lane * 8;
    float4 va = *(const float4*)row;
    float4 vb = *(const float4*)(row + 4);
    float mx = fmaxf(fmaxf(fmaxf(va.x, va.y), fmaxf(va.z, va.w)),
                     fmaxf(fmaxf(vb.x, vb.y), fmaxf(vb.z, vb.w)));
    #pragma unroll
    for (int off = 16; off; off >>= 1) mx = fmaxf(mx, __shfl_xor(mx, off, 32));
    va.x = __expf(va.x - mx); va.y = __expf(va.y - mx);
    va.z = __expf(va.z - mx); va.w = __expf(va.w - mx);
    vb.x = __expf(vb.x - mx); vb.y = __expf(vb.y - mx);
    vb.z = __expf(vb.z - mx); vb.w = __expf(vb.w - mx);
    float s = va.x + va.y + va.z + va.w + vb.x + vb.y + vb.z + vb.w;
    #pragma unroll
    for (int off = 16; off; off >>= 1) s += __shfl_xor(s, off, 32);
    const float inv = 1.0f / s;
    va.x *= inv; va.y *= inv; va.z *= inv; va.w *= inv;
    vb.x *= inv; vb.y *= inv; vb.z *= inv; vb.w *= inv;
    *(float4*)row = va; *(float4*)(row + 4) = vb;
  }
  __syncthreads();

  // ---- O = P @ V: per q-tile, 2 accumulators share one A fragment ----
  for (int tq = wv; tq < 16; tq += 8) {
    const int qrow = tq * 16 + lo;
    v8f acc0 = {}, acc1 = {};
    for (int kb = 0; kb < 256; kb += 32) {
      Frag a, b0, b1;
      a.q[0] = pack8(&Ls[qrow * 256 + kb + hi * 8]);
      a.q[1] = pack8(&Ls[qrow * 256 + kb + 16 + hi * 8]);
      b0.q[0] = *(const uint4*)&Vt[lo * 256 + kb + hi * 8];
      b0.q[1] = *(const uint4*)&Vt[lo * 256 + kb + 16 + hi * 8];
      b1.q[0] = *(const uint4*)&Vt[(16 + lo) * 256 + kb + hi * 8];
      b1.q[1] = *(const uint4*)&Vt[(16 + lo) * 256 + kb + 16 + hi * 8];
      acc0 = wmma_bf16(a.v, b0.v, acc0);
      acc1 = wmma_bf16(a.v, b1.v, acc1);
    }
    #pragma unroll
    for (int j = 0; j < 8; ++j) {
      const int r = tq * 16 + hi * 8 + j;
      O[base + r * RS + lo]      = f2bfu(acc0[j]);
      O[base + r * RS + 16 + lo] = f2bfu(acc1[j]);
    }
  }
}

// ---------------------------------------------------------------------------
// bias[h,i,j] = sum_c pair[i,j,c] * Wb[c,h] + bb[h]   (CZ=128, tiny GEMV)
// ---------------------------------------------------------------------------
__global__ __launch_bounds__(256)
void pair_bias_kernel(const float* __restrict__ pair, const float* __restrict__ Wb,
                      const float* __restrict__ bb, float* __restrict__ bias, int nh) {
  const int idx = blockIdx.x * 256 + threadIdx.x;
  if (idx >= nh * 65536) return;
  const int h = idx >> 16, ij = idx & 65535;
  const float* p = pair + (long)ij * 128;
  float a = bb[h];
  for (int c = 0; c < 128; ++c) a += p[c] * Wb[c * nh + h];
  bias[idx] = a;                                        // layout [h][i][j]
}

// ---------------------------------------------------------------------------
// LayerNorm: one wave per row of length C; lanes own contiguous chunks.
// ---------------------------------------------------------------------------
__global__ __launch_bounds__(256)
void ln_kernel(const float* __restrict__ X, const float* __restrict__ g,
               const float* __restrict__ b, float* __restrict__ Y, int C, int rows) {
  const int wid = blockIdx.x * 8 + (threadIdx.x >> 5);
  if (wid >= rows) return;
  const int lane = threadIdx.x & 31;
  const int n = C >> 5;                                 // 8 (C=256) or 4 (C=128)
  const float* x = X + (long)wid * C + lane * n;
  float v[8], s = 0.0f;
  for (int i = 0; i < n; ++i) { v[i] = x[i]; s += v[i]; }
  for (int off = 16; off; off >>= 1) s += __shfl_xor(s, off, 32);
  const float m = s / (float)C;
  float q = 0.0f;
  for (int i = 0; i < n; ++i) { const float d = v[i] - m; q += d * d; }
  for (int off = 16; off; off >>= 1) q += __shfl_xor(q, off, 32);
  const float inv = rsqrtf(q / (float)C + 1e-5f);
  float* y = Y + (long)wid * C + lane * n;
  for (int i = 0; i < n; ++i) {
    const int c = lane * n + i;
    y[i] = (v[i] - m) * inv * g[c] + b[c];
  }
}

// mean over the sequence axis: mean[r,c] = (1/256) sum_s msa[s,r,c]
__global__ __launch_bounds__(256)
void mean_kernel(const float* __restrict__ msa, float* __restrict__ mean) {
  const int idx = blockIdx.x * 256 + threadIdx.x;       // idx = r*256 + c
  if (idx >= 65536) return;
  float a = 0.0f;
  for (int s = 0; s < 256; ++s) a += msa[(long)s * 65536 + idx];
  mean[idx] = a * (1.0f / 256.0f);
}

// ---------------------------------------------------------------------------
// Outer-product-mean fused per (i,j): op = mean[i]*mean[j]; LN; @W1(256,32)
// relu; @W2(32,128); pair_out = pair_in + result. 256-thread block per ij.
// ---------------------------------------------------------------------------
__global__ __launch_bounds__(256)
void opm_kernel(const float* __restrict__ mean, const float* __restrict__ g,
                const float* __restrict__ b, const float* __restrict__ W1,
                const float* __restrict__ b1, const float* __restrict__ W2,
                const float* __restrict__ b2, const float* __restrict__ pair_in,
                float* __restrict__ pair_out) {
  __shared__ float red[256];
  __shared__ float hln[256];
  __shared__ float h1s[32];
  const int ij = blockIdx.x, i = ij >> 8, j = ij & 255;
  const int t = threadIdx.x;
  const float op = mean[i * 256 + t] * mean[j * 256 + t];
  red[t] = op; __syncthreads();
  for (int s = 128; s; s >>= 1) { if (t < s) red[t] += red[t + s]; __syncthreads(); }
  const float m = red[0] * (1.0f / 256.0f); __syncthreads();
  const float d = op - m;
  red[t] = d * d; __syncthreads();
  for (int s = 128; s; s >>= 1) { if (t < s) red[t] += red[t + s]; __syncthreads(); }
  const float inv = rsqrtf(red[0] * (1.0f / 256.0f) + 1e-5f);
  hln[t] = d * inv * g[t] + b[t];
  __syncthreads();
  if (t < 32) {
    float a = b1[t];
    for (int c = 0; c < 256; ++c) a += hln[c] * W1[c * 32 + t];
    h1s[t] = fmaxf(a, 0.0f);
  }
  __syncthreads();
  if (t < 128) {
    float a = b2[t];
    for (int o = 0; o < 32; ++o) a += h1s[o] * W2[o * 128 + t];
    pair_out[(long)ij * 128 + t] = pair_in[(long)ij * 128 + t] + a;
  }
}

// ---------------------------------------------------------------------------
// Host orchestration
// ---------------------------------------------------------------------------
extern "C" void kernel_launch(void* const* d_in, const int* in_sizes, int n_in,
                              void* d_out, int out_size, void* d_ws, size_t ws_size,
                              hipStream_t stream) {
  (void)in_sizes; (void)n_in; (void)out_size; (void)ws_size;
  enum { ROW_WB = 2, ROW_BB, ROW_WQ, ROW_BQ, ROW_WK, ROW_BK, ROW_WV, ROW_BV,
         ROW_WO, ROW_BO, ROW_LNG, ROW_LNB,
         COL_WB, COL_BB, COL_WQ, COL_BQ, COL_WK, COL_BK, COL_WV, COL_BV,
         COL_WO, COL_BO, COL_LNG, COL_LNB,
         MT_LNG, MT_LNB, MT_W1, MT_B1, MT_W2, MT_B2,
         OPM_LNG, OPM_LNB, OPM_W1, OPM_B1, OPM_W2, OPM_B2,
         TRI_LNG, TRI_LNB, TRI_WQ, TRI_BQ, TRI_WK, TRI_BK, TRI_WV, TRI_BV,
         TRI_WO, TRI_BO,
         PT_LNG, PT_LNB, PT_W1, PT_B1, PT_W2, PT_B2 };
  const float* msa0  = (const float*)d_in[0];
  const float* pair0 = (const float*)d_in[1];
  auto P = [&](int i) { return (const float*)d_in[i]; };

  const long MROW = 65536;                              // S*R = 256*256
  char* ws = (char*)d_ws; size_t off = 0;
  auto alloc = [&](size_t bytes) -> char* {
    char* p = ws + off; off += (bytes + 255) & ~(size_t)255; return p;
  };
  float*          M1  = (float*)alloc(MROW * 256 * 4);
  float*          M0  = (float*)alloc(MROW * 256 * 4);
  unsigned short* Qb  = (unsigned short*)alloc(MROW * 256 * 2);
  unsigned short* Kb  = (unsigned short*)alloc(MROW * 256 * 2);
  unsigned short* Vb  = (unsigned short*)alloc(MROW * 256 * 2);
  unsigned short* Ob  = (unsigned short*)alloc(MROW * 256 * 2);
  unsigned short* Hb  = (unsigned short*)alloc(MROW * 1024 * 2);
  float*          BI  = (float*)alloc((size_t)8 * 65536 * 4);
  float*          MEA = (float*)alloc((size_t)65536 * 4);
  float*          PR  = (float*)alloc(MROW * 128 * 4);
  float*          P1  = (float*)alloc(MROW * 128 * 4);

  float* msa_out  = (float*)d_out;                      // 16,777,216 floats
  float* pair_out = msa_out + 16777216;                 //  8,388,608 floats

  const size_t asmem = 3 * 256 * 32 * sizeof(unsigned short) + 256 * 256 * sizeof(float);
  (void)hipFuncSetAttribute((const void*)attn_kernel<true>,
                            hipFuncAttributeMaxDynamicSharedMemorySize, (int)asmem);
  (void)hipFuncSetAttribute((const void*)attn_kernel<false>,
                            hipFuncAttributeMaxDynamicSharedMemorySize, (int)asmem);

  auto ggrid = [](long M, long N) { return (int)((M >> 6) * (N >> 6)); };

  // ======== Row attention ========
  pair_bias_kernel<<<(8 * 65536) / 256, 256, 0, stream>>>(pair0, P(ROW_WB), P(ROW_BB), BI, 8);
  gemm_kernel<false, true, false, false><<<ggrid(MROW, 256), 256, 0, stream>>>(msa0, P(ROW_WQ), P(ROW_BQ), nullptr, Qb, MROW, 256, 256);
  gemm_kernel<false, true, false, false><<<ggrid(MROW, 256), 256, 0, stream>>>(msa0, P(ROW_WK), P(ROW_BK), nullptr, Kb, MROW, 256, 256);
  gemm_kernel<false, true, false, false><<<ggrid(MROW, 256), 256, 0, stream>>>(msa0, P(ROW_WV), P(ROW_BV), nullptr, Vb, MROW, 256, 256);
  attn_kernel<true><<<256 * 8, 256, asmem, stream>>>(Qb, Kb, Vb, Ob, BI, 8, 65536, 256);
  gemm_kernel<true, false, false, true><<<ggrid(MROW, 256), 256, 0, stream>>>(Ob, P(ROW_WO), P(ROW_BO), msa0, M1, MROW, 256, 256);
  ln_kernel<<<MROW / 8, 256, 0, stream>>>(M1, P(ROW_LNG), P(ROW_LNB), M0, 256, MROW);

  // ======== Column attention (same buffers, column strides) ========
  pair_bias_kernel<<<(8 * 65536) / 256, 256, 0, stream>>>(pair0, P(COL_WB), P(COL_BB), BI, 8);
  gemm_kernel<false, true, false, false><<<ggrid(MROW, 256), 256, 0, stream>>>(M0, P(COL_WQ), P(COL_BQ), nullptr, Qb, MROW, 256, 256);
  gemm_kernel<false, true, false, false><<<ggrid(MROW, 256), 256, 0, stream>>>(M0, P(COL_WK), P(COL_BK), nullptr, Kb, MROW, 256, 256);
  gemm_kernel<false, true, false, false><<<ggrid(MROW, 256), 256, 0, stream>>>(M0, P(COL_WV), P(COL_BV), nullptr, Vb, MROW, 256, 256);
  attn_kernel<true><<<256 * 8, 256, asmem, stream>>>(Qb, Kb, Vb, Ob, BI, 8, 256, 65536);
  gemm_kernel<true, false, false, true><<<ggrid(MROW, 256), 256, 0, stream>>>(Ob, P(COL_WO), P(COL_BO), M0, M1, MROW, 256, 256);
  ln_kernel<<<MROW / 8, 256, 0, stream>>>(M1, P(COL_LNG), P(COL_LNB), M0, 256, MROW);

  // ======== MSA transition ========
  ln_kernel<<<MROW / 8, 256, 0, stream>>>(M0, P(MT_LNG), P(MT_LNB), M1, 256, MROW);
  gemm_kernel<false, true, true, false><<<ggrid(MROW, 1024), 256, 0, stream>>>(M1, P(MT_W1), P(MT_B1), nullptr, Hb, MROW, 1024, 256);
  gemm_kernel<true, false, false, true><<<ggrid(MROW, 256), 256, 0, stream>>>(Hb, P(MT_W2), P(MT_B2), M0, msa_out, MROW, 256, 1024);

  // ======== Outer-product-mean ========
  mean_kernel<<<256, 256, 0, stream>>>(msa_out, MEA);
  opm_kernel<<<65536, 256, 0, stream>>>(MEA, P(OPM_LNG), P(OPM_LNB), P(OPM_W1), P(OPM_B1),
                                        P(OPM_W2), P(OPM_B2), pair0, PR);

  // ======== Triangle attention (4 heads, hd=32, no bias) ========
  ln_kernel<<<MROW / 8, 256, 0, stream>>>(PR, P(TRI_LNG), P(TRI_LNB), P1, 128, MROW);
  gemm_kernel<false, true, false, false><<<ggrid(MROW, 128), 256, 0, stream>>>(P1, P(TRI_WQ), P(TRI_BQ), nullptr, Qb, MROW, 128, 128);
  gemm_kernel<false, true, false, false><<<ggrid(MROW, 128), 256, 0, stream>>>(P1, P(TRI_WK), P(TRI_BK), nullptr, Kb, MROW, 128, 128);
  gemm_kernel<false, true, false, false><<<ggrid(MROW, 128), 256, 0, stream>>>(P1, P(TRI_WV), P(TRI_BV), nullptr, Vb, MROW, 128, 128);
  attn_kernel<false><<<256 * 4, 256, asmem, stream>>>(Qb, Kb, Vb, Ob, nullptr, 4, 32768, 128);
  gemm_kernel<true, false, false, true><<<ggrid(MROW, 128), 256, 0, stream>>>(Ob, P(TRI_WO), P(TRI_BO), PR, P1, MROW, 128, 128);

  // ======== Pair transition ========
  ln_kernel<<<MROW / 8, 256, 0, stream>>>(P1, P(PT_LNG), P(PT_LNB), PR, 128, MROW);
  gemm_kernel<false, true, true, false><<<ggrid(MROW, 512), 256, 0, stream>>>(PR, P(PT_W1), P(PT_B1), nullptr, Hb, MROW, 512, 128);
  gemm_kernel<true, false, false, true><<<ggrid(MROW, 128), 256, 0, stream>>>(Hb, P(PT_W2), P(PT_B2), P1, pair_out, MROW, 128, 512);
}
